// TemporalModel_61272003444756
// MI455X (gfx1250) — matrix operand order repaired and checked by reading, not compiled, not run on hardware
//
#include <hip/hip_runtime.h>
#include <hip/hip_bf16.h>

// ---------------------------------------------------------------------------
// Types for CDNA5 WMMA (wave32, V_WMMA_F32_16X16X32_F16)
// ---------------------------------------------------------------------------
typedef __attribute__((ext_vector_type(16))) _Float16 v16h;
typedef __attribute__((ext_vector_type(8)))  _Float16 v8h;
typedef __attribute__((ext_vector_type(8)))  float    v8f;

#define Bn 32
#define Tn 128
#define NC 64
#define DK 128

__device__ __forceinline__ float sigmoidf_(float x) {
    return 1.0f / (1.0f + __expf(-x));
}

__device__ __forceinline__ v8f wmma32(v16h a, v16h b, v8f c) {
    // D = A(16x32 f16) * B(32x16 f16) + C(16x16 f32)
    return __builtin_amdgcn_wmma_f32_16x16x32_f16(
        /*neg_a=*/false, a, /*neg_b=*/false, b,
        /*c_mod=*/(short)0, c, /*reuse_a=*/false, /*reuse_b=*/false);
}

// B fragment for k-tile kt. Bmat[k][n] = Wrow[k] where Wrow = row n of the
// (row-major) weight matrix, i.e. B = W^T. Layout (dense 16-bit B 32x16):
// lanes 0-15 (N=lane) hold K {0..7,16..23}; lanes 16-31 (N=lane-16) hold
// K {8..15,24..31} -> halves 0..7 = Wrow[k0+j], 8..15 = Wrow[k0+16+j], k0 = kt*32 + (hi?8:0).
__device__ __forceinline__ v16h load_bfrag(const float* wrow, int kt, int lane) {
    int k0 = kt * 32 + ((lane & 16) ? 8 : 0);
    v16h b;
#pragma unroll
    for (int j = 0; j < 8; ++j) { b[j] = (_Float16)wrow[k0 + j]; b[j + 8] = (_Float16)wrow[k0 + 16 + j]; }
    return b;
}

// A fragment for k-tile kt from a contiguous f16 row (128 K values).
// Layout (16-bit A 16x32): lanes 0-15 (M=lane) hold K {0..7,16..23};
// lanes 16-31 (M=lane-16) hold K {8..15,24..31}.
// halves 0..7 = K k0..k0+7, halves 8..15 = K k0+16..k0+23, k0 = kt*32+(hi?8:0)
__device__ __forceinline__ v16h load_afrag_row(const _Float16* row, int kt, int lane) {
    v16h a;
    int k0 = kt * 32 + ((lane & 16) ? 8 : 0);
    if (row) {
        v8h lo = *(const v8h*)(row + k0);
        v8h hi = *(const v8h*)(row + k0 + 16);
#pragma unroll
        for (int j = 0; j < 8; ++j) { a[j] = lo[j]; a[j + 8] = hi[j]; }
    } else {
#pragma unroll
        for (int j = 0; j < 16; ++j) a[j] = (_Float16)0.0f;
    }
    return a;
}

// ---------------------------------------------------------------------------
// Kernel 1: gathers + f16 staging + small weight preprocessing
// ---------------------------------------------------------------------------
__global__ __launch_bounds__(256) void prep_kernel(
    const int* __restrict__ q_data, const float* __restrict__ q_embed_w,
    const float* __restrict__ W1, const float* __restrict__ W2,
    const float* __restrict__ W3, const float* __restrict__ W4,
    const float* __restrict__ W5,
    float* __restrict__ qemb_out, _Float16* __restrict__ Ea16,
    float* __restrict__ w1sum,
    _Float16* __restrict__ w2c, _Float16* __restrict__ w3c,
    _Float16* __restrict__ w4b, _Float16* __restrict__ w5b)
{
    int idx = blockIdx.x * 256 + threadIdx.x;   // 0 .. 4096*128-1
    int r = idx >> 7;
    int d = idx & 127;
    float v = q_embed_w[(size_t)q_data[r] * DK + d];
    qemb_out[idx] = v;
    Ea16[idx] = (_Float16)v;

    if (idx < 128) {
        float s = 0.0f;
#pragma unroll 4
        for (int j = 0; j < 128; ++j) s += W1[idx * 256 + 128 + j];
        w1sum[idx] = s;
    }
    if (idx < 4 * 128 * 128) {
        int which = idx >> 14;
        int loc   = idx & 16383;
        int dd    = loc >> 7;
        int k     = loc & 127;
        float wv;
        _Float16* dst;
        switch (which) {
            case 0:  wv = W2[dd * 384 + 256 + k]; dst = w2c; break;
            case 1:  wv = W3[dd * 384 + 256 + k]; dst = w3c; break;
            case 2:  wv = W4[dd * 256 + 128 + k]; dst = w4b; break;
            default: wv = W5[dd * 256 + 128 + k]; dst = w5b; break;
        }
        dst[loc] = (_Float16)wv;
    }
}

// ---------------------------------------------------------------------------
// Kernel 2: qa GEMM.  qa[r,:] = q_emb[r,:] @ W1a^T + a[r]*w1sum + b1
// ---------------------------------------------------------------------------
__global__ __launch_bounds__(256) void gemm_qa_kernel(
    const _Float16* __restrict__ Ea16, const float* __restrict__ W1,
    const float* __restrict__ b1, const float* __restrict__ w1sum,
    const float* __restrict__ a_data,
    float* __restrict__ qa_out, _Float16* __restrict__ Qa16)
{
    const int mBase = blockIdx.x * 64;
    const int lane  = threadIdx.x & 31;
    const int wv    = threadIdx.x >> 5;
    const int n     = wv * 16 + (lane & 15);

    v8f acc[4];
#pragma unroll
    for (int mt = 0; mt < 4; ++mt) acc[mt] = (v8f)(0.0f);

#pragma unroll
    for (int kt = 0; kt < 4; ++kt) {
        v16h bf = load_bfrag(W1 + (size_t)n * 256, kt, lane);
#pragma unroll
        for (int mt = 0; mt < 4; ++mt) {
            int r = mBase + mt * 16 + (lane & 15);
            v16h a = load_afrag_row(Ea16 + (size_t)r * DK, kt, lane);
            acc[mt] = wmma32(a, bf, acc[mt]);
        }
    }
    float ws = w1sum[n];
    float bb = b1[n];
#pragma unroll
    for (int mt = 0; mt < 4; ++mt) {
#pragma unroll
        for (int vr = 0; vr < 8; ++vr) {
            int r = mBase + mt * 16 + vr + ((lane & 16) ? 8 : 0);
            float v = acc[mt][vr] + a_data[r] * ws + bb;
            qa_out[(size_t)r * DK + n] = v;
            Qa16[(size_t)r * DK + n]   = (_Float16)v;
        }
    }
}

// ---------------------------------------------------------------------------
// Kernel 3: G2/G3 GEMM (K=256, shared A fragments, dual accumulators)
// ---------------------------------------------------------------------------
__global__ __launch_bounds__(256) void gemm_g23_kernel(
    const _Float16* __restrict__ Qa16,
    const float* __restrict__ W2, const float* __restrict__ W3,
    const float* __restrict__ b2, const float* __restrict__ b3,
    float* __restrict__ G2, float* __restrict__ G3)
{
    const int mBase = blockIdx.x * 64;
    const int lane  = threadIdx.x & 31;
    const int wv    = threadIdx.x >> 5;
    const int n     = wv * 16 + (lane & 15);

    v8f acc2[4], acc3[4];
#pragma unroll
    for (int mt = 0; mt < 4; ++mt) { acc2[mt] = (v8f)(0.0f); acc3[mt] = (v8f)(0.0f); }

#pragma unroll
    for (int kt = 0; kt < 8; ++kt) {
        v16h bf2 = load_bfrag(W2 + (size_t)n * 384, kt, lane);
        v16h bf3 = load_bfrag(W3 + (size_t)n * 384, kt, lane);
#pragma unroll
        for (int mt = 0; mt < 4; ++mt) {
            int r = mBase + mt * 16 + (lane & 15);
            const _Float16* row;
            int kk;
            if (kt < 4) {  // learning_pre half
                row = ((r & 127) == 0) ? (const _Float16*)nullptr
                                       : (Qa16 + (size_t)(r - 1) * DK);
                kk = kt;
            } else {       // learning half
                row = Qa16 + (size_t)r * DK;
                kk = kt - 4;
            }
            v16h a = load_afrag_row(row, kk, lane);
            acc2[mt] = wmma32(a, bf2, acc2[mt]);
            acc3[mt] = wmma32(a, bf3, acc3[mt]);
        }
    }
    float bb2 = b2[n], bb3 = b3[n];
#pragma unroll
    for (int mt = 0; mt < 4; ++mt) {
#pragma unroll
        for (int vr = 0; vr < 8; ++vr) {
            int r = mBase + mt * 16 + vr + ((lane & 16) ? 8 : 0);
            G2[(size_t)r * DK + n] = acc2[mt][vr] + bb2;
            G3[(size_t)r * DK + n] = acc3[mt][vr] + bb3;
        }
    }
}

// ---------------------------------------------------------------------------
// Kernel 4: G5 GEMM.  G5[r,:] = W5[:, :128] @ q_emb[b, t+1] + b5
// ---------------------------------------------------------------------------
__global__ __launch_bounds__(256) void gemm_g5_kernel(
    const _Float16* __restrict__ Ea16, const float* __restrict__ W5,
    const float* __restrict__ b5, float* __restrict__ G5)
{
    const int mBase = blockIdx.x * 64;
    const int lane  = threadIdx.x & 31;
    const int wv    = threadIdx.x >> 5;
    const int n     = wv * 16 + (lane & 15);

    v8f acc[4];
#pragma unroll
    for (int mt = 0; mt < 4; ++mt) acc[mt] = (v8f)(0.0f);

#pragma unroll
    for (int kt = 0; kt < 4; ++kt) {
        v16h bf = load_bfrag(W5 + (size_t)n * 256, kt, lane);
#pragma unroll
        for (int mt = 0; mt < 4; ++mt) {
            int r  = mBase + mt * 16 + (lane & 15);
            int rr = (r >= 4095) ? 4095 : (r + 1);   // q_emb at t+1
            v16h a = load_afrag_row(Ea16 + (size_t)rr * DK, kt, lane);
            acc[mt] = wmma32(a, bf, acc[mt]);
        }
    }
    float bb = b5[n];
#pragma unroll
    for (int mt = 0; mt < 4; ++mt) {
#pragma unroll
        for (int vr = 0; vr < 8; ++vr) {
            int r = mBase + mt * 16 + vr + ((lane & 16) ? 8 : 0);
            G5[(size_t)r * DK + n] = acc[mt][vr] + bb;
        }
    }
}

// ---------------------------------------------------------------------------
// Kernel 5: recurrence. One block (8 wave32s) per batch element.
// State h[64][128] in LDS (f32 master + f16 WMMA copy). All four recurrent
// GEMV weight blocks (W2c/W3c/W4b/W5b, f16) are ALSO resident in LDS
// (loaded once, padded to 136-half rows for conflict-free ds_read_b128),
// removing ~128 KB/step of L2 traffic from the serial critical path.
// Per step: P = h_pre @ W4a^T via 16 v_wmma per wave; W4a^T B-fragments are
// loop-invariant in registers.
// ---------------------------------------------------------------------------
#define S32 129   // padded f32 row stride (floats) -> conflict-free column reads
#define S16 136   // padded f16 row stride (halves) -> conflict-free row reads
#define WMAT (128 * S16)   // one 128x128 f16 weight block in LDS (padded)

__global__ __launch_bounds__(256) void recur_kernel(
    const int* __restrict__ q_data, const float* __restrict__ q_matrix,
    const float* __restrict__ W4, const float* __restrict__ b4,
    const float* __restrict__ h0,
    const _Float16* __restrict__ w2c, const _Float16* __restrict__ w3c,
    const _Float16* __restrict__ w4b, const _Float16* __restrict__ w5b,
    const float* __restrict__ G2, const float* __restrict__ G3,
    const float* __restrict__ G5,
    float* __restrict__ Hout, float* __restrict__ pred)
{
    __shared__ float h32[NC * S32];
    __shared__ _Float16 __attribute__((aligned(16))) h16[NC * S16];
    __shared__ _Float16 __attribute__((aligned(16))) wL[4 * WMAT]; // w2c|w3c|w4b|w5b
    __shared__ float ht[DK], KP[DK], Rr[DK], KC[DK], Uv[DK];
    __shared__ float g2s[DK], g3s[DK], g5s[DK], red[DK], b4s[DK];
    __shared__ float qe[NC], qen[NC];

    const int b    = blockIdx.x;
    const int tid  = threadIdx.x;
    const int lane = tid & 31;
    const int wv   = tid >> 5;

    // ---- init: h = h0 (broadcast), weights -> LDS, pred[:,0]=0, h_tilde0 ----
    for (int i = tid; i < NC * DK; i += 256) {
        int n = i >> 7, d = i & 127;
        float v = h0[i];
        h32[n * S32 + d] = v;
        h16[n * S16 + d] = (_Float16)v;
    }
    for (int i = tid; i < 4 * 128 * 128; i += 256) {
        int mat = i >> 14;
        int loc = i & 16383;
        int d   = loc >> 7;
        int k   = loc & 127;
        const _Float16* src = (mat == 0) ? w2c : (mat == 1) ? w3c
                            : (mat == 2) ? w4b : w5b;
        wL[mat * WMAT + d * S16 + k] = src[loc];
    }
    if (tid < DK) b4s[tid] = b4[tid];
    if (tid < NC) qe[tid] = q_matrix[(size_t)q_data[b * Tn] * NC + tid];
    if (tid == 0) pred[b * Tn] = 0.0f;
    __syncthreads();
    if (tid < DK) {
        float acc = 0.0f;
        for (int n = 0; n < NC; ++n) acc += qe[n] * h32[n * S32 + tid];
        ht[tid] = acc;   // h_tilde0 = qm[:,0] @ h0
    }

    // ---- loop-invariant B fragments of W4a^T (B[k][n] = W4[n][k], k<128) ----
    const int dcol = wv * 16 + (lane & 15);   // this lane's output column d
    v16h bfr[4];
#pragma unroll
    for (int kt = 0; kt < 4; ++kt)
        bfr[kt] = load_bfrag(W4 + (size_t)dcol * 256, kt, lane);
    __syncthreads();

    for (int t = 0; t < Tn - 1; ++t) {
        // (1) H[t] = sigmoid(h_pre)  -> global
        {
            size_t base = (size_t)(t * Bn + b) * (NC * DK);
            for (int i = tid; i < NC * DK; i += 256) {
                int n = i >> 7, d = i & 127;
                Hout[base + i] = sigmoidf_(h32[n * S32 + d]);
            }
        }
        // (2) stage per-step vectors (+ prefetch next step's rows into caches)
        if (tid < DK) {
            size_t r = (size_t)(b * Tn + t) * DK;
            g2s[tid] = G2[r + tid];
            g3s[tid] = G3[r + tid];
            g5s[tid] = G5[r + tid];
            if (t + 1 < Tn - 1) {
                __builtin_prefetch(G2 + r + DK + tid, 0, 0);
                __builtin_prefetch(G3 + r + DK + tid, 0, 0);
                __builtin_prefetch(G5 + r + DK + tid, 0, 0);
            }
        } else if (tid < 192) {
            int n = tid - 128;
            qe[n]  = q_matrix[(size_t)q_data[b * Tn + t] * NC + n];
        } else {
            int n = tid - 192;
            qen[n] = q_matrix[(size_t)q_data[b * Tn + t + 1] * NC + n];
        }
        __syncthreads();

        // (3) KP = tanh(G2 + W2c@ht) ; R = sigmoid(G3 + W3c@ht)   [LDS f16]
        if (tid < DK) {
            const v8h* wp = (const v8h*)(wL + 0 * WMAT + tid * S16);
            float acc = g2s[tid];
#pragma unroll 4
            for (int kk = 0; kk < 16; ++kk) {
                v8h wvv = wp[kk];
#pragma unroll
                for (int j = 0; j < 8; ++j) acc += (float)wvv[j] * ht[kk * 8 + j];
            }
            KP[tid] = tanhf(acc);
        } else {
            int d = tid - DK;
            const v8h* wp = (const v8h*)(wL + 1 * WMAT + d * S16);
            float acc = g3s[d];
#pragma unroll 4
            for (int kk = 0; kk < 16; ++kk) {
                v8h wvv = wp[kk];
#pragma unroll
                for (int j = 0; j < 8; ++j) acc += (float)wvv[j] * ht[kk * 8 + j];
            }
            Rr[d] = sigmoidf_(acc);
        }
        __syncthreads();
        if (tid < DK) KC[tid] = Rr[tid] * (KP[tid] + 1.0f) * 0.5f;
        __syncthreads();

        // (4) U = b4 + W4b @ KC                                   [LDS f16]
        if (tid < DK) {
            const v8h* wp = (const v8h*)(wL + 2 * WMAT + tid * S16);
            float acc = b4s[tid];
#pragma unroll 4
            for (int kk = 0; kk < 16; ++kk) {
                v8h wvv = wp[kk];
#pragma unroll
                for (int j = 0; j < 8; ++j) acc += (float)wvv[j] * KC[kk * 8 + j];
            }
            Uv[tid] = acc;
        }
        __syncthreads();

        // (5) P = h_pre @ W4a^T : 4 m-tiles x 4 k-tiles of v_wmma per wave
        v8f acc[4];
#pragma unroll
        for (int mt = 0; mt < 4; ++mt) {
            v8f c = (v8f)(0.0f);
#pragma unroll
            for (int kt = 0; kt < 4; ++kt) {
                int m  = mt * 16 + (lane & 15);
                int k0 = kt * 32 + ((lane & 16) ? 8 : 0);
                v8h lo = *(const v8h*)(h16 + m * S16 + k0);
                v8h hi = *(const v8h*)(h16 + m * S16 + k0 + 16);
                v16h a;
#pragma unroll
                for (int j = 0; j < 8; ++j) { a[j] = lo[j]; a[j + 8] = hi[j]; }
                c = wmma32(a, bfr[kt], c);
            }
            acc[mt] = c;
        }
        __syncthreads();   // all h16 A-reads done before state update

        // (6) epilogue: h_new[n][d] = qe[n]*KC[d] + sigmoid(P+U[d])*h_old[n][d]
        {
            float u  = Uv[dcol];
            float kc = KC[dcol];
#pragma unroll
            for (int mt = 0; mt < 4; ++mt) {
#pragma unroll
                for (int vr = 0; vr < 8; ++vr) {
                    int n = mt * 16 + vr + ((lane & 16) ? 8 : 0);
                    float hold = h32[n * S32 + dcol];
                    float hnew = qe[n] * kc + sigmoidf_(acc[mt][vr] + u) * hold;
                    h32[n * S32 + dcol] = hnew;
                    h16[n * S16 + dcol] = (_Float16)hnew;
                }
            }
        }
        __syncthreads();

        // (7) h_tilde = qm[:,t+1] @ h_new
        if (tid < DK) {
            float a2 = 0.0f;
            for (int n = 0; n < NC; ++n) a2 += qen[n] * h32[n * S32 + tid];
            ht[tid] = a2;
        }
        __syncthreads();

        // (8) y = mean(sigmoid(G5 + W5b @ h_tilde))               [LDS f16]
        if (tid < DK) {
            const v8h* wp = (const v8h*)(wL + 3 * WMAT + tid * S16);
            float acc2 = g5s[tid];
#pragma unroll 4
            for (int kk = 0; kk < 16; ++kk) {
                v8h wvv = wp[kk];
#pragma unroll
                for (int j = 0; j < 8; ++j) acc2 += (float)wvv[j] * ht[kk * 8 + j];
            }
            red[tid] = sigmoidf_(acc2);
        }
        __syncthreads();
        if (tid < 64) red[tid] += red[tid + 64];
        __syncthreads();
        if (tid < 32) {
            float s = red[tid] + red[tid + 32];
#pragma unroll
            for (int o = 16; o > 0; o >>= 1) s += __shfl_down(s, o, 32);
            if (tid == 0) pred[b * Tn + t + 1] = s * (1.0f / 128.0f);
        }
        __syncthreads();
    }
}

// ---------------------------------------------------------------------------
// Host launcher
// ---------------------------------------------------------------------------
extern "C" void kernel_launch(void* const* d_in, const int* in_sizes, int n_in,
                              void* d_out, int out_size, void* d_ws, size_t ws_size,
                              hipStream_t stream) {
    const int*   q_data    = (const int*)  d_in[0];
    const float* a_data    = (const float*)d_in[1];
    const float* q_matrix  = (const float*)d_in[2];
    const float* q_embed_w = (const float*)d_in[3];
    const float* W1 = (const float*)d_in[4];
    const float* b1 = (const float*)d_in[5];
    const float* W2 = (const float*)d_in[6];
    const float* b2 = (const float*)d_in[7];
    const float* W3 = (const float*)d_in[8];
    const float* b3 = (const float*)d_in[9];
    const float* W4 = (const float*)d_in[10];
    const float* b4 = (const float*)d_in[11];
    const float* W5 = (const float*)d_in[12];
    const float* b5 = (const float*)d_in[13];
    const float* h0 = (const float*)d_in[14];

    // Output layout: H [127,32,64,128] | pred [32,128] | q_emb [32,128,128] | qa
    float* out   = (float*)d_out;
    float* Hout  = out;
    float* pred  = out + (size_t)127 * 32 * 64 * 128;     // 33,292,288
    float* qemb  = pred + 32 * 128;                       // +4096
    float* qa    = qemb + (size_t)32 * 128 * 128;         // +524,288

    // Workspace layout (bytes)
    char* ws = (char*)d_ws;
    _Float16* Ea16 = (_Float16*)(ws + 0);          // 4096*128 f16 = 1 MB
    _Float16* Qa16 = (_Float16*)(ws + 1048576);    // 1 MB
    float* G2      = (float*)(ws + 2097152);       // 2 MB
    float* G3      = (float*)(ws + 4194304);       // 2 MB
    float* G5      = (float*)(ws + 6291456);       // 2 MB
    float* w1sum   = (float*)(ws + 8388608);       // 512 B
    _Float16* w2c  = (_Float16*)(ws + 8389120);    // 32 KB each
    _Float16* w3c  = w2c + 128 * 128;
    _Float16* w4b  = w3c + 128 * 128;
    _Float16* w5b  = w4b + 128 * 128;

    prep_kernel<<<2048, 256, 0, stream>>>(q_data, q_embed_w, W1, W2, W3, W4, W5,
                                          qemb, Ea16, w1sum, w2c, w3c, w4b, w5b);
    gemm_qa_kernel<<<64, 256, 0, stream>>>(Ea16, W1, b1, w1sum, a_data, qa, Qa16);
    gemm_g23_kernel<<<64, 256, 0, stream>>>(Qa16, W2, W3, b2, b3, G2, G3);
    gemm_g5_kernel<<<64, 256, 0, stream>>>(Ea16, W5, b5, G5);
    recur_kernel<<<32, 256, 0, stream>>>(q_data, q_matrix, W4, b4, h0,
                                         w2c, w3c, w4b, w5b, G2, G3, G5,
                                         Hout, pred);
}
